// PairScorer_1975684956405
// MI455X (gfx1250) — compile-verified, matrix operand order checked by reading, not run on hardware
//
#include <hip/hip_runtime.h>
#include <hip/hip_bf16.h>

typedef __attribute__((ext_vector_type(16))) __bf16 v16bf;
typedef __attribute__((ext_vector_type(8)))  float  v8f;

#define N_SPAN 1024
#define WIN    250
#define DEMB   20
#define HDIM   150
#define NPAIR  (N_SPAN * WIN)      // 256000
#define KF     416                 // padded feature dim (13 * 32)
#define KH     160                 // padded hidden dim (5 * 32, 10 * 16)
#define ROWS   64                  // pairs per block-iteration
#define NBLK   (NPAIR / ROWS)      // 4000
#define NKT1   13                  // k-tiles layer 1
#define NKT2   5                   // k-tiles layer 2/3
#define NCT    10                  // col tiles (160/16)

// ---- LDS byte offsets (all 16B aligned) ----
#define OFF_W1  0                          // 130 frags * 1024B
#define OFF_W2  133120                     // 50 frags * 1024B
#define OFF_W3  184320                     // 5 frags * 1024B
#define OFF_B1  189440                     // 160 f32
#define OFF_B2  190080                     // 160 f32
#define OFF_X   190720                     // 64 x 416 bf16
#define OFF_H1  243968                     // 64 x 160 bf16
#define OFF_H2  264448                     // 64 x 160 bf16
#define OFF_SA  284928                     // 64 f32
#define LDS_BYTES 285184

__device__ __forceinline__ unsigned short f2bf(float f) {
    unsigned u = __builtin_bit_cast(unsigned, f);
    unsigned r = u + 0x7FFFu + ((u >> 16) & 1u);   // round-to-nearest-even
    return (unsigned short)(r >> 16);
}

// A-operand fragment: row-major bf16 rows in LDS -> documented 16x32 bf16 A layout.
__device__ __forceinline__ v16bf load_a(const unsigned short* rows, int rt, int kt,
                                        int lane, int pitch) {
    int r    = rt * 16 + (lane & 15);
    int grp8 = (lane >> 4) * 8;
    const unsigned short* p = rows + r * pitch + kt * 32 + grp8;
    union { uint4 u[2]; v16bf v; } c;
    c.u[0] = *(const uint4*)(p);
    c.u[1] = *(const uint4*)(p + 16);
    return c.v;
}

// B-operand fragment: pre-swizzled in LDS, 32B contiguous per lane.
__device__ __forceinline__ v16bf load_b(const unsigned short* frag, int lane) {
    union { uint4 u[2]; v16bf v; } c;
    const uint4* p = (const uint4*)(frag + lane * 16);
    c.u[0] = p[0];
    c.u[1] = p[1];
    return c.v;
}

// K index inside a 32-deep tile for fragment element (lane, slot): slot=2*v+half.
__device__ __forceinline__ int frag_k(int lane, int slot) {
    int v = slot >> 1, half = slot & 1;
    return ((v >= 4) ? 16 : 0) + (lane >> 4) * 8 + (v & 3) * 2 + half;
}

__global__ void __launch_bounds__(256)
pair_scorer_wmma(const float* __restrict__ g, const int* __restrict__ i1,
                 const float* __restrict__ sm, const float* __restrict__ dist_table,
                 const float* __restrict__ W1, const float* __restrict__ b1,
                 const float* __restrict__ W2, const float* __restrict__ b2,
                 const float* __restrict__ W3, const float* __restrict__ b3,
                 float* __restrict__ out) {
    extern __shared__ char smem[];
    unsigned short* smW1 = (unsigned short*)(smem + OFF_W1);
    unsigned short* smW2 = (unsigned short*)(smem + OFF_W2);
    unsigned short* smW3 = (unsigned short*)(smem + OFF_W3);
    float*          smB1 = (float*)(smem + OFF_B1);
    float*          smB2 = (float*)(smem + OFF_B2);
    unsigned short* smX  = (unsigned short*)(smem + OFF_X);
    unsigned short* smH1 = (unsigned short*)(smem + OFF_H1);
    unsigned short* smH2 = (unsigned short*)(smem + OFF_H2);
    float*          smSA = (float*)(smem + OFF_SA);

    const int tid  = threadIdx.x;
    const int lane = tid & 31;
    const int wave = tid >> 5;

    // ---- epsilon column of sij: zero every launch ----
    for (int i = blockIdx.x * 256 + tid; i < N_SPAN; i += gridDim.x * 256)
        out[i * (WIN + 1) + WIN] = 0.0f;

    // ---- stage weights into LDS, pre-swizzled into B-fragment layout ----
    for (int idx = tid; idx < NCT * NKT1 * 512; idx += 256) {     // W1: 130 frags
        int f = idx >> 9, e = idx & 511;
        int ln = e >> 4, slot = e & 15;
        int ct = f / NKT1, kt = f % NKT1;
        int k = kt * 32 + frag_k(ln, slot);
        int n = ct * 16 + (ln & 15);
        float w = (k < 3 * 128 + DEMB && n < HDIM) ? W1[k * HDIM + n] : 0.0f;
        smW1[f * 512 + ln * 16 + slot] = f2bf(w);
    }
    for (int idx = tid; idx < NCT * NKT2 * 512; idx += 256) {     // W2: 50 frags
        int f = idx >> 9, e = idx & 511;
        int ln = e >> 4, slot = e & 15;
        int ct = f / NKT2, kt = f % NKT2;
        int k = kt * 32 + frag_k(ln, slot);
        int n = ct * 16 + (ln & 15);
        float w = (k < HDIM && n < HDIM) ? W2[k * HDIM + n] : 0.0f;
        smW2[f * 512 + ln * 16 + slot] = f2bf(w);
    }
    for (int idx = tid; idx < NKT2 * 512; idx += 256) {           // W3 in column 0 only
        int f = idx >> 9, e = idx & 511;
        int ln = e >> 4, slot = e & 15;
        int k = f * 32 + frag_k(ln, slot);
        float w = (((ln & 15) == 0) && k < HDIM) ? W3[k] : 0.0f;
        smW3[f * 512 + ln * 16 + slot] = f2bf(w);
    }
    for (int idx = tid; idx < KH; idx += 256) {
        smB1[idx] = (idx < HDIM) ? b1[idx] : 0.0f;
        smB2[idx] = (idx < HDIM) ? b2[idx] : 0.0f;
    }
    __syncthreads();

    const float b3v = b3[0];
    const int rt    = wave >> 1;      // row tile owned by this wave (0..3)
    const int chalf = wave & 1;       // which 5 column tiles (0..1)
    const int nloc  = lane & 15;
    const int rbase0 = rt * 16 + (lane >> 4) * 8;

    // ---- grid-stride over 64-pair blocks ----
    for (int blk = blockIdx.x; blk < NBLK; blk += gridDim.x) {
        const int row0 = blk * ROWS;

        // Build 64x416 bf16 feature rows, float4 at a time.
        for (int idx = tid; idx < ROWS * (KF / 4); idx += 256) {
            int r  = idx / (KF / 4);
            int c  = (idx - r * (KF / 4)) * 4;
            int p  = row0 + r;
            int i  = p / WIN, w = p - i * WIN;
            int j  = i - WIN + w;
            int js = (j < 0) ? 0 : j;
            float4 x;
            if (c < 128)       x = *(const float4*)(g + i * 128 + c);
            else if (c < 256)  x = *(const float4*)(g + js * 128 + (c - 128));
            else if (c < 384) {
                float4 xa = *(const float4*)(g + i * 128 + (c - 256));
                float4 xb = *(const float4*)(g + js * 128 + (c - 256));
                x = make_float4(xa.x * xb.x, xa.y * xb.y, xa.z * xb.z, xa.w * xb.w);
            } else if (c < 384 + DEMB) {
                int d = i1[i] - i1[js];
                int didx = (d >= 1) + (d >= 2) + (d >= 3) + (d >= 4) +
                           (d >= 8) + (d >= 16) + (d >= 32) + (d >= 64);
                x = *(const float4*)(dist_table + didx * DEMB + (c - 384));
            } else x = make_float4(0.f, 0.f, 0.f, 0.f);
            uint2 pk;
            pk.x = (unsigned)f2bf(x.x) | ((unsigned)f2bf(x.y) << 16);
            pk.y = (unsigned)f2bf(x.z) | ((unsigned)f2bf(x.w) << 16);
            *(uint2*)(smX + r * KF + c) = pk;
        }
        __syncthreads();

        // ---- Layer 1: 5 accumulators per wave, double-buffered fragments ----
        {
            v8f acc[5] = {};
            v16bf a[2], b[2][5];
            a[0] = load_a(smX, rt, 0, lane, KF);
#pragma unroll
            for (int u = 0; u < 5; ++u)
                b[0][u] = load_b(smW1 + ((chalf * 5 + u) * NKT1) * 512, lane);
#pragma unroll
            for (int kt = 0; kt < NKT1; ++kt) {
                int cur = kt & 1, nxt = cur ^ 1;
                if (kt + 1 < NKT1) {
                    a[nxt] = load_a(smX, rt, kt + 1, lane, KF);
#pragma unroll
                    for (int u = 0; u < 5; ++u)
                        b[nxt][u] = load_b(smW1 + ((chalf * 5 + u) * NKT1 + kt + 1) * 512, lane);
                }
#pragma unroll
                for (int u = 0; u < 5; ++u)
                    acc[u] = __builtin_amdgcn_wmma_f32_16x16x32_bf16(
                        false, a[cur], false, b[cur][u], (short)0, acc[u], false, false);
            }
#pragma unroll
            for (int u = 0; u < 5; ++u) {
                int ncol = (chalf * 5 + u) * 16 + nloc;
                float bias = smB1[ncol];
#pragma unroll
                for (int v = 0; v < 8; ++v) {
                    float h = acc[u][v] + bias;
                    smH1[(rbase0 + v) * KH + ncol] = f2bf(h > 0.0f ? h : 0.0f);
                }
            }
        }
        __syncthreads();

        // ---- Layer 2: same structure, K = 160 ----
        {
            v8f acc[5] = {};
            v16bf a[2], b[2][5];
            a[0] = load_a(smH1, rt, 0, lane, KH);
#pragma unroll
            for (int u = 0; u < 5; ++u)
                b[0][u] = load_b(smW2 + ((chalf * 5 + u) * NKT2) * 512, lane);
#pragma unroll
            for (int kt = 0; kt < NKT2; ++kt) {
                int cur = kt & 1, nxt = cur ^ 1;
                if (kt + 1 < NKT2) {
                    a[nxt] = load_a(smH1, rt, kt + 1, lane, KH);
#pragma unroll
                    for (int u = 0; u < 5; ++u)
                        b[nxt][u] = load_b(smW2 + ((chalf * 5 + u) * NKT2 + kt + 1) * 512, lane);
                }
#pragma unroll
                for (int u = 0; u < 5; ++u)
                    acc[u] = __builtin_amdgcn_wmma_f32_16x16x32_bf16(
                        false, a[cur], false, b[cur][u], (short)0, acc[u], false, false);
            }
#pragma unroll
            for (int u = 0; u < 5; ++u) {
                int ncol = (chalf * 5 + u) * 16 + nloc;
                float bias = smB2[ncol];
#pragma unroll
                for (int v = 0; v < 8; ++v) {
                    float h = acc[u][v] + bias;
                    smH2[(rbase0 + v) * KH + ncol] = f2bf(h > 0.0f ? h : 0.0f);
                }
            }
        }
        __syncthreads();

        // ---- Layer 3: [64 x 160] @ [160 x 16] (W3 in col 0) -> smSA ----
        if (wave < 4) {
            int rt3 = wave;
            v8f acc = {};
#pragma unroll
            for (int kt = 0; kt < NKT2; ++kt) {
                v16bf a = load_a(smH2, rt3, kt, lane, KH);
                v16bf b = load_b(smW3 + kt * 512, lane);
                acc = __builtin_amdgcn_wmma_f32_16x16x32_bf16(false, a, false, b,
                                                              (short)0, acc, false, false);
            }
            if ((lane & 15) == 0) {       // lanes 0 and 16 hold column 0
                int rb = rt3 * 16 + (lane >> 4) * 8;
#pragma unroll
                for (int v = 0; v < 8; ++v) smSA[rb + v] = acc[v];
            }
        }
        __syncthreads();

        // ---- Epilogue: sij = mask ? sm_i + sm_j + sa + b3 : 0 ; mask as float ----
        for (int r = tid; r < ROWS; r += 256) {
            int p = row0 + r;
            int i = p / WIN, w = p - i * WIN;
            int j = i - WIN + w;
            float val = 0.0f, m = 0.0f;
            if (j >= 0) {
                val = sm[i] + sm[j] + smSA[r] + b3v;
                m = 1.0f;
            }
            out[i * (WIN + 1) + w] = val;
            out[N_SPAN * (WIN + 1) + i * WIN + w] = m;
        }
        __syncthreads();   // protect LDS reuse by next block-iteration
    }
}

extern "C" void kernel_launch(void* const* d_in, const int* in_sizes, int n_in,
                              void* d_out, int out_size, void* d_ws, size_t ws_size,
                              hipStream_t stream) {
    const float* g          = (const float*)d_in[0];
    const int*   i1         = (const int*)  d_in[1];
    const float* sm         = (const float*)d_in[2];
    const float* dist_table = (const float*)d_in[3];
    const float* W1         = (const float*)d_in[4];
    const float* b1         = (const float*)d_in[5];
    const float* W2         = (const float*)d_in[6];
    const float* b2         = (const float*)d_in[7];
    const float* W3         = (const float*)d_in[8];
    const float* b3         = (const float*)d_in[9];
    float*       out        = (float*)d_out;

    dim3 grid(1024), block(256);
    pair_scorer_wmma<<<grid, block, LDS_BYTES, stream>>>(
        g, i1, sm, dist_table, W1, b1, W2, b2, W3, b3, out);
}
